// TinyGPT_49950469653176
// MI455X (gfx1250) — matrix operand order repaired and checked by reading, not compiled, not run on hardware
//
#include <hip/hip_runtime.h>
#include <hip/hip_bf16.h>

typedef __attribute__((ext_vector_type(16))) __bf16 v16bf;
typedef __attribute__((ext_vector_type(8)))  float  v8f;
typedef __attribute__((ext_vector_type(4)))  int    v4i;

constexpr int Bc = 2, Tc = 2048, Dc = 1024, Hc = 16, DHc = 64;
constexpr int DFFc = 4096, Lc = 8, NVc = 256, TTEXTc = 1792, VOCABc = 32000;

enum { F_RELU = 1, F_RES = 2, F_ROWMAP = 4, F_QKV = 8, F_OUTF = 16, F_OUTB = 32 };

// ---- CDNA5 async global->LDS copy (ASYNCcnt path), with safe fallback -----
#if defined(__has_builtin)
#if __has_builtin(__builtin_amdgcn_global_load_async_to_lds_b128) && \
    __has_builtin(__builtin_amdgcn_s_wait_asynccnt)
#define HAVE_ASYNC_LDS 1
#endif
#endif

#if defined(HAVE_ASYNC_LDS)
typedef __attribute__((address_space(1))) v4i gv4i;   // global int4
typedef __attribute__((address_space(3))) v4i lv4i;   // LDS int4
__device__ __forceinline__ void async_copy16(const void* g, void* l) {
  __builtin_amdgcn_global_load_async_to_lds_b128((gv4i*)g, (lv4i*)l, 0, 0);
}
__device__ __forceinline__ void async_wait0() {
  __builtin_amdgcn_s_wait_asynccnt(0);
}
#else
__device__ __forceinline__ void async_copy16(const void* g, void* l) {
  *(uint4*)l = *(const uint4*)g;
}
__device__ __forceinline__ void async_wait0() {}
#endif

__device__ __forceinline__ unsigned short f2bf(float f) {
  unsigned int u = __float_as_uint(f);
  u += 0x7FFFu + ((u >> 16) & 1u);           // round-to-nearest-even
  return (unsigned short)(u >> 16);
}

union Frag {
  v16bf v;
  unsigned short h[16];
  unsigned int   u[8];
};

__device__ __forceinline__ v8f wmma_bf16(const Frag& a, const Frag& b, v8f c) {
  return __builtin_amdgcn_wmma_f32_16x16x32_bf16(false, a.v, false, b.v,
                                                 (short)0, c, false, false);
}

// ---------------------------------------------------------------------------
// x[b,t,d] = concat(image_embeds, tok_emb[idx])[b,t,d] + pos_emb[t,d]
// ---------------------------------------------------------------------------
__global__ void k_embed(const int* __restrict__ idx,
                        const float* __restrict__ img,
                        const float* __restrict__ tok,
                        const float* __restrict__ pos,
                        float* __restrict__ x) {
  const long long n = (long long)Bc * Tc * Dc;
  long long i = (long long)blockIdx.x * blockDim.x + threadIdx.x;
  const long long stride = (long long)gridDim.x * blockDim.x;
  for (; i < n; i += stride) {
    int d = (int)(i % Dc);
    long long bt = i / Dc;
    int t = (int)(bt % Tc);
    int b = (int)(bt / Tc);
    float base;
    if (t < NVc) base = img[((long long)b * NVc + t) * Dc + d];
    else         base = tok[(long long)idx[b * TTEXTc + (t - NVc)] * Dc + d];
    x[i] = base + pos[(long long)t * Dc + d];
  }
}

// ---------------------------------------------------------------------------
// LayerNorm: one 256-thread block per row (D=1024), bf16 output for WMMA.
// ---------------------------------------------------------------------------
__global__ __launch_bounds__(256) void k_ln(const float* __restrict__ x,
                                            const float* __restrict__ w,
                                            const float* __restrict__ bias,
                                            unsigned short* __restrict__ out) {
  const int row = blockIdx.x;
  const int tid = threadIdx.x;
  const float* xr = x + (size_t)row * Dc;
  float v[4];
#pragma unroll
  for (int i = 0; i < 4; ++i) v[i] = xr[tid + i * 256];

  __shared__ float sred[8];
  float s = v[0] + v[1] + v[2] + v[3];
#pragma unroll
  for (int m = 16; m >= 1; m >>= 1) s += __shfl_xor(s, m, 32);
  if ((tid & 31) == 0) sred[tid >> 5] = s;
  __syncthreads();
  float tot = 0.f;
#pragma unroll
  for (int i = 0; i < 8; ++i) tot += sred[i];
  const float mu = tot * (1.0f / Dc);
  __syncthreads();

  float sq = 0.f;
#pragma unroll
  for (int i = 0; i < 4; ++i) { float d = v[i] - mu; sq += d * d; }
#pragma unroll
  for (int m = 16; m >= 1; m >>= 1) sq += __shfl_xor(sq, m, 32);
  if ((tid & 31) == 0) sred[tid >> 5] = sq;
  __syncthreads();
  float vtot = 0.f;
#pragma unroll
  for (int i = 0; i < 8; ++i) vtot += sred[i];
  const float rstd = rsqrtf(vtot * (1.0f / Dc) + 1e-5f);

  unsigned short* outr = out + (size_t)row * Dc;
#pragma unroll
  for (int i = 0; i < 4; ++i) {
    int c = tid + i * 256;
    outr[c] = f2bf((v[i] - mu) * rstd * w[c] + bias[c]);
  }
}

// ---------------------------------------------------------------------------
// GEMM: out[M,N] = A[M,K](bf16) @ W[N,K]^T (fp32 weights, converted in LDS).
// 128x128 tile per block (256 thr = 8 waves, wave tile 64x32, BK=32).
// A staged row-major (async path); W staged TRANSPOSED [k][n] so B-fragment
// reads are contiguous across lanes (n = nb + m16).
// ---------------------------------------------------------------------------
__global__ __launch_bounds__(256) void k_gemm(
    const unsigned short* __restrict__ A, const float* __restrict__ Wf,
    const float* __restrict__ bias, float* outf, unsigned short* outb,
    int M, int N, int K, int flags, float qscale,
    unsigned short* qb, unsigned short* kbuf, unsigned short* vb) {
  __shared__ __align__(16) unsigned short As[128][32];    // [m][k]
  __shared__ __align__(16) unsigned short Ws[32][128];    // [k][n]  (transposed)

  const int tid = threadIdx.x;
  const int bn = blockIdx.x, bm = blockIdx.y;
  const int wave = tid >> 5, lane = tid & 31;
  const int wm = wave >> 2, wn = wave & 3;       // 2 x 4 wave grid
  const int m16 = lane & 15, g = lane >> 4;

  v8f acc[4][2];
#pragma unroll
  for (int mi = 0; mi < 4; ++mi)
#pragma unroll
    for (int ni = 0; ni < 2; ++ni)
#pragma unroll
      for (int r = 0; r < 8; ++r) acc[mi][ni][r] = 0.0f;

  for (int k0 = 0; k0 < K; k0 += 32) {
    // Stage A (bf16, 128x32): async global->LDS b128 when available
#pragma unroll
    for (int it = 0; it < 2; ++it) {
      int e = tid + it * 256;
      int r = e >> 2, q = e & 3;
      int lrow = bm * 128 + r;
      int grow = lrow;
      if (flags & F_ROWMAP) {                    // skip vision rows for head
        int bb = lrow / TTEXTc;
        grow = bb * Tc + NVc + (lrow - bb * TTEXTc);
      }
      const unsigned short* ap = A + (size_t)grow * K + k0 + q * 8;
      async_copy16(ap, &As[r][q * 8]);
      if (it == 0 && k0 + 32 < K) __builtin_prefetch(ap + 32, 0, 3);
    }
    // Stage W (fp32 -> bf16, transposed into Ws[k][n])
#pragma unroll
    for (int it = 0; it < 4; ++it) {
      int e = tid + it * 256;
      int r = e >> 3, q = e & 7;                 // r = n row, q*4 = k group
      int nrow = bn * 128 + r;
      const float* wp = Wf + (size_t)nrow * K + k0 + q * 4;
      float4 wv = *(const float4*)wp;
      Ws[q * 4 + 0][r] = f2bf(wv.x);
      Ws[q * 4 + 1][r] = f2bf(wv.y);
      Ws[q * 4 + 2][r] = f2bf(wv.z);
      Ws[q * 4 + 3][r] = f2bf(wv.w);
      if (it == 0 && k0 + 32 < K) __builtin_prefetch(wp + 32, 0, 3);
    }
    async_wait0();
    __syncthreads();

    // A fragments: lane = M row, 16 halves over K (ISA 16-bit A layout)
    Frag a[4];
#pragma unroll
    for (int mi = 0; mi < 4; ++mi) {
      int row = wm * 64 + mi * 16 + m16;
#pragma unroll
      for (int j = 0; j < 8; ++j) {
        int kk = (j < 4) ? (g * 8 + 2 * j) : (8 + g * 8 + 2 * j);
        a[mi].u[j] = *(const unsigned int*)&As[row][kk];
      }
    }
    // B fragments: lane = N col, halves over K (ISA 16-bit B layout)
#pragma unroll
    for (int ni = 0; ni < 2; ++ni) {
      Frag bf;
      int nb = wn * 32 + ni * 16;
#pragma unroll
      for (int j = 0; j < 8; ++j) {
        int kk = (j < 4) ? (g * 8 + 2 * j) : (8 + g * 8 + 2 * j);
        bf.u[j] = (unsigned int)Ws[kk][nb + m16] | ((unsigned int)Ws[kk + 1][nb + m16] << 16);
      }
#pragma unroll
      for (int mi = 0; mi < 4; ++mi)
        acc[mi][ni] = wmma_bf16(a[mi], bf, acc[mi][ni]);
    }
    __syncthreads();
  }

  // Epilogue
#pragma unroll
  for (int mi = 0; mi < 4; ++mi)
#pragma unroll
    for (int ni = 0; ni < 2; ++ni) {
      int n = bn * 128 + wn * 32 + ni * 16 + m16;
      float bv = bias ? bias[n] : 0.0f;
#pragma unroll
      for (int r = 0; r < 8; ++r) {
        int m = bm * 128 + wm * 64 + mi * 16 + r + 8 * g;
        float val = acc[mi][ni][r] + bv;
        if (flags & F_RELU) val = fmaxf(val, 0.0f);
        if (flags & F_QKV) {
          int which = n >> 10, d = n & 1023;
          int hh = d >> 6, dh = d & 63;
          int bb = m >> 11, t = m & (Tc - 1);
          unsigned short* dst = (which == 0) ? qb : (which == 1 ? kbuf : vb);
          float sv = (which == 0) ? val * qscale : val;
          dst[(((size_t)bb * Hc + hh) * Tc + t) * DHc + dh] = f2bf(sv);
        } else {
          size_t oi = (size_t)m * N + n;
          if (flags & F_RES)  val += outf[oi];
          if (flags & F_OUTF) outf[oi] = val;
          if (flags & F_OUTB) outb[oi] = f2bf(val);
        }
      }
    }
}

// ---------------------------------------------------------------------------
// Flash attention over [B,H,T,DH] bf16 Q(pre-scaled)/K/V; prefix mask:
// blocked <=> (q >= NV && k > q). Block = 128 q rows, 8 waves x 16 rows.
// K staged TRANSPOSED [d][key] for contiguous B-fragment reads; V staged
// row-major via async copy.
// ---------------------------------------------------------------------------
__global__ __launch_bounds__(256) void k_attn(const unsigned short* __restrict__ Q,
                                              const unsigned short* __restrict__ Kg,
                                              const unsigned short* __restrict__ Vg,
                                              unsigned short* __restrict__ obf) {
  __shared__ __align__(16) unsigned short KtT[64][32];   // [d][key]
  __shared__ __align__(16) unsigned short Vt[32][64];    // [key][d]
  __shared__ __align__(16) unsigned short Ps[8][16][32];

  const int tid = threadIdx.x;
  const int bh = blockIdx.y;
  const int b = bh >> 4, h = bh & 15;
  const int qbase = blockIdx.x * 128;
  const int wave = tid >> 5, lane = tid & 31;
  const int m16 = lane & 15, g = lane >> 4;

  // Q A-fragments (two K=32 chunks over DH=64)
  const int qrow = qbase + wave * 16 + m16;
  const unsigned short* Qp = Q + ((size_t)bh * Tc + qrow) * DHc;
  Frag aQ[2];
#pragma unroll
  for (int dc = 0; dc < 2; ++dc)
#pragma unroll
    for (int j = 0; j < 8; ++j) {
      int kk = dc * 32 + ((j < 4) ? (g * 8 + 2 * j) : (8 + g * 8 + 2 * j));
      aQ[dc].u[j] = *(const unsigned int*)(Qp + kk);
    }

  v8f o[4];
#pragma unroll
  for (int ni = 0; ni < 4; ++ni)
#pragma unroll
    for (int r = 0; r < 8; ++r) o[ni][r] = 0.0f;
  float mrow[8], lrow[8];
#pragma unroll
  for (int r = 0; r < 8; ++r) { mrow[r] = -1e30f; lrow[r] = 0.0f; }

  const unsigned short* Kbase = Kg + (size_t)bh * Tc * DHc;
  const unsigned short* Vbase = Vg + (size_t)bh * Tc * DHc;
  const int kend = (qbase >= NVc) ? (qbase + 128) : Tc;

  for (int kb0 = 0; kb0 < kend; kb0 += 32) {
    {   // stage K (transposed, manual) and V (row-major, async) tiles
      int rr = tid >> 3, qq = tid & 7;          // rr = key row, qq*8 = d group
      uint4 kv = *(const uint4*)(Kbase + (size_t)(kb0 + rr) * DHc + qq * 8);
      const unsigned short* ks = (const unsigned short*)&kv;
#pragma unroll
      for (int j = 0; j < 8; ++j) KtT[qq * 8 + j][rr] = ks[j];
      async_copy16(Vbase + (size_t)(kb0 + rr) * DHc + qq * 8, &Vt[rr][qq * 8]);
    }
    async_wait0();
    __syncthreads();

    // S = Q K^T for 32 keys: two 16x16 C-frags, K-dim = DH in two chunks
    v8f s0, s1;
#pragma unroll
    for (int r = 0; r < 8; ++r) { s0[r] = 0.0f; s1[r] = 0.0f; }
#pragma unroll
    for (int dc = 0; dc < 2; ++dc) {
      Frag bk0, bk1;
#pragma unroll
      for (int j = 0; j < 8; ++j) {
        int kk = dc * 32 + ((j < 4) ? (g * 8 + 2 * j) : (8 + g * 8 + 2 * j));
        bk0.u[j] = (unsigned int)KtT[kk][m16] | ((unsigned int)KtT[kk + 1][m16] << 16);
        bk1.u[j] = (unsigned int)KtT[kk][16 + m16] | ((unsigned int)KtT[kk + 1][16 + m16] << 16);
      }
      s0 = wmma_bf16(aQ[dc], bk0, s0);
      s1 = wmma_bf16(aQ[dc], bk1, s1);
    }

    // online softmax (row stats live per-lane for the 8 rows of this half)
#pragma unroll
    for (int r = 0; r < 8; ++r) {
      int qg = qbase + wave * 16 + r + 8 * g;
      float v0 = s0[r];
      if (qg >= NVc && (kb0 + m16) > qg) v0 = -1e30f;
      float v1 = s1[r];
      if (qg >= NVc && (kb0 + 16 + m16) > qg) v1 = -1e30f;
      float mx = fmaxf(v0, v1);
#pragma unroll
      for (int sh = 8; sh >= 1; sh >>= 1) mx = fmaxf(mx, __shfl_xor(mx, sh, 32));
      float mnew = fmaxf(mrow[r], mx);
      float e0 = __expf(v0 - mnew), e1 = __expf(v1 - mnew);
      Ps[wave][r + 8 * g][m16]      = f2bf(e0);
      Ps[wave][r + 8 * g][16 + m16] = f2bf(e1);
      float ps = e0 + e1;
#pragma unroll
      for (int sh = 8; sh >= 1; sh >>= 1) ps += __shfl_xor(ps, sh, 32);
      float alpha = __expf(mrow[r] - mnew);
      lrow[r] = lrow[r] * alpha + ps;
      mrow[r] = mnew;
#pragma unroll
      for (int ni = 0; ni < 4; ++ni) o[ni][r] *= alpha;
    }

    // P (C-layout) -> A-layout via per-wave LDS scratch, then O += P @ V
    Frag p;
#pragma unroll
    for (int j = 0; j < 8; ++j) {
      int kk = (j < 4) ? (g * 8 + 2 * j) : (8 + g * 8 + 2 * j);
      p.u[j] = *(const unsigned int*)&Ps[wave][m16][kk];
    }
#pragma unroll
    for (int ni = 0; ni < 4; ++ni) {
      Frag bv;
#pragma unroll
      for (int j = 0; j < 8; ++j) {
        int kk = (j < 4) ? (g * 8 + 2 * j) : (8 + g * 8 + 2 * j);
        bv.u[j] = (unsigned int)Vt[kk][ni * 16 + m16] | ((unsigned int)Vt[kk + 1][ni * 16 + m16] << 16);
      }
      o[ni] = wmma_bf16(p, bv, o[ni]);
    }
    __syncthreads();
  }

  // write merged heads: obf[b, t, h*64 + col]
#pragma unroll
  for (int ni = 0; ni < 4; ++ni)
#pragma unroll
    for (int r = 0; r < 8; ++r) {
      int t = qbase + wave * 16 + r + 8 * g;
      int col = h * DHc + ni * 16 + m16;
      obf[((size_t)b * Tc + t) * Dc + col] = f2bf(o[ni][r] / lrow[r]);
    }
}

// ---------------------------------------------------------------------------
extern "C" void kernel_launch(void* const* d_in, const int* in_sizes, int n_in,
                              void* d_out, int out_size, void* d_ws, size_t ws_size,
                              hipStream_t stream) {
  (void)in_sizes; (void)n_in; (void)out_size; (void)ws_size;
  const int*   idx   = (const int*)d_in[0];
  const float* img   = (const float*)d_in[1];
  const float* tok   = (const float*)d_in[2];
  const float* pos   = (const float*)d_in[3];
  const float* ln1w  = (const float*)d_in[4];
  const float* ln1b  = (const float*)d_in[5];
  const float* qkvw  = (const float*)d_in[6];
  const float* outw  = (const float*)d_in[7];
  const float* ln2w  = (const float*)d_in[8];
  const float* ln2b  = (const float*)d_in[9];
  const float* fc1w  = (const float*)d_in[10];
  const float* fc1b  = (const float*)d_in[11];
  const float* fc2w  = (const float*)d_in[12];
  const float* fc2b  = (const float*)d_in[13];
  const float* lnfw  = (const float*)d_in[14];
  const float* lnfb  = (const float*)d_in[15];
  const float* headw = (const float*)d_in[16];
  float* logits = (float*)d_out;

  char* ws = (char*)d_ws;
  size_t off = 0;
  auto alloc = [&](size_t bytes) -> char* {
    off = (off + 255) & ~(size_t)255;
    char* p = ws + off;
    off += bytes;
    return p;
  };
  float*          x   = (float*)alloc((size_t)Bc * Tc * Dc * 4);
  unsigned short* xb  = (unsigned short*)alloc((size_t)Bc * Tc * Dc * 2);
  unsigned short* obf = (unsigned short*)alloc((size_t)Bc * Tc * Dc * 2);
  unsigned short* qb  = (unsigned short*)alloc((size_t)Bc * Hc * Tc * DHc * 2);
  unsigned short* kb  = (unsigned short*)alloc((size_t)Bc * Hc * Tc * DHc * 2);
  unsigned short* vb  = (unsigned short*)alloc((size_t)Bc * Hc * Tc * DHc * 2);
  unsigned short* ffh = (unsigned short*)alloc((size_t)Bc * Tc * DFFc * 2);

  const int rows = Bc * Tc;                       // 4096
  const float qscale = 0.125f;                    // 1/sqrt(DH)

  k_embed<<<4096, 256, 0, stream>>>(idx, img, tok, pos, x);

  for (int l = 0; l < Lc; ++l) {
    const float* wq  = qkvw + (size_t)l * 3 * Dc * Dc;
    const float* wo  = outw + (size_t)l * Dc * Dc;
    const float* w1  = fc1w + (size_t)l * DFFc * Dc;
    const float* w2  = fc2w + (size_t)l * Dc * DFFc;

    k_ln<<<rows, 256, 0, stream>>>(x, ln1w + l * Dc, ln1b + l * Dc, xb);

    // QKV projection, scatter to [B,H,T,DH] (Q pre-scaled)
    k_gemm<<<dim3(3 * Dc / 128, rows / 128), 256, 0, stream>>>(
        xb, wq, nullptr, nullptr, nullptr, rows, 3 * Dc, Dc,
        F_QKV, qscale, qb, kb, vb);

    k_attn<<<dim3(Tc / 128, Bc * Hc), 256, 0, stream>>>(qb, kb, vb, obf);

    // out projection + residual into x
    k_gemm<<<dim3(Dc / 128, rows / 128), 256, 0, stream>>>(
        obf, wo, nullptr, x, nullptr, rows, Dc, Dc,
        F_RES | F_OUTF, 0.f, nullptr, nullptr, nullptr);

    k_ln<<<rows, 256, 0, stream>>>(x, ln2w + l * Dc, ln2b + l * Dc, xb);

    // fc1 + bias + relu -> bf16 hidden
    k_gemm<<<dim3(DFFc / 128, rows / 128), 256, 0, stream>>>(
        xb, w1, fc1b + l * DFFc, nullptr, ffh, rows, DFFc, Dc,
        F_RELU | F_OUTB, 0.f, nullptr, nullptr, nullptr);

    // fc2 + bias + residual into x
    k_gemm<<<dim3(Dc / 128, rows / 128), 256, 0, stream>>>(
        ffh, w2, fc2b + l * Dc, x, nullptr, rows, Dc, DFFc,
        F_RES | F_OUTF, 0.f, nullptr, nullptr, nullptr);
  }

  k_ln<<<rows, 256, 0, stream>>>(x, lnfw, lnfb, xb);

  // head over text rows only: logits[B,TTEXT,VOCAB]
  k_gemm<<<dim3(VOCABc / 128, (Bc * TTEXTc) / 128), 256, 0, stream>>>(
      xb, headw, nullptr, logits, nullptr, Bc * TTEXTc, VOCABc, Dc,
      F_ROWMAP | F_OUTF, 0.f, nullptr, nullptr, nullptr);
}
